// AME2Encoder_28235115004072
// MI455X (gfx1250) — compile-verified, matrix operand order checked by reading, not run on hardware
//
#include <hip/hip_runtime.h>
#include <hip/hip_bf16.h>

typedef _Float16 h16;
typedef __attribute__((ext_vector_type(16))) _Float16 v16h;
typedef __attribute__((ext_vector_type(8)))  _Float16 v8h;
typedef __attribute__((ext_vector_type(8)))  float    v8f;
typedef unsigned int uint32x4 __attribute__((ext_vector_type(4)));
typedef int          int32x4  __attribute__((ext_vector_type(4)));
typedef int          int32x8  __attribute__((ext_vector_type(8)));

#define HH 14
#define WWD 36
#define LL 504
#define LP 512
// weight-pack region offsets, in halves (B-fragment swizzled layout)
#define OFF_C1 0        // conv1 im2col  K=32  N=64   (nkt=1, nt=4)  2048
#define OFF_C2 2048     // conv2         K=64  N=64   (nkt=2, nt=4)  4096
#define OFF_FU 6144     // fuse (top)    K=64  N=64   (nkt=2, nt=4)  4096
#define OFF_G1 10240    // g_w1          K=64  N=128  (nkt=2, nt=8)  8192
#define OFF_G2 18432    // g_w2          K=128 N=128  (nkt=4, nt=8) 16384
#define OFF_WK 34816    // wk            K=64  N=64   (nkt=2, nt=4)  4096
#define OFF_WV 38912    // wv            K=64  N=64   (nkt=2, nt=4)  4096
#define WPACK_HALVES 43008
#define WPACK_BYTES (WPACK_HALVES * 2)
#define PETERM_BYTES (LP * 64 * 4)

#if defined(__has_builtin)
#if __has_builtin(__builtin_amdgcn_tensor_load_to_lds) && \
    __has_builtin(__builtin_amdgcn_s_wait_tensorcnt)
#define USE_TDM 1
#endif
#endif

__device__ __forceinline__ float eluf(float x) {
  return x > 0.f ? x : __expf(x) - 1.f;
}

__device__ __forceinline__ v8f wmma32(v16h a, v16h b, v8f c) {
  // D = A(16x32,f16) * B(32x16,f16) + C(16x16,f32)
  return __builtin_amdgcn_wmma_f32_16x16x32_f16(false, a, false, b, (short)0, c,
                                                false, false);
}

// A fragment from row-major f16 tile in LDS (stride in halves).
// lane<16: M=lane, K=0..7 & 16..23 ; lane>=16: M=lane-16, K=8..15 & 24..31
__device__ __forceinline__ v16h load_Afrag(const h16* tile, int stride, int lane) {
  int m = lane & 15, kb = (lane >> 4) << 3;
  const h16* p = tile + m * stride + kb;
  v8h lo = *(const v8h*)p;
  v8h hi = *(const v8h*)(p + 16);
  v16h a;
#pragma unroll
  for (int e = 0; e < 8; ++e) { a[e] = lo[e]; a[e + 8] = hi[e]; }
  return a;
}

// B fragment from pre-swizzled weight pack: 512 halves per (nt,kt) block,
// each lane's 16 elements contiguous -> two ds_load_b128.
__device__ __forceinline__ v16h load_Bfrag(const h16* mat, int nt, int nkt,
                                           int kt, int lane) {
  const h16* p = mat + (((nt * nkt + kt) * 32 + lane) << 4);
  v8h lo = *(const v8h*)p;
  v8h hi = *(const v8h*)(p + 8);
  v16h b;
#pragma unroll
  for (int e = 0; e < 8; ++e) { b[e] = lo[e]; b[e + 8] = hi[e]; }
  return b;
}

// C init: broadcast bias[n] to all 8 accumulator elements (N fixed per lane)
__device__ __forceinline__ v8f bias_C(const float* bias, int n0, int lane) {
  float v = bias[n0 + (lane & 15)];
  v8f c;
#pragma unroll
  for (int i = 0; i < 8; ++i) c[i] = v;
  return c;
}

// Store D (f32 accum) to row-major f16 LDS tile, optional ELU.
// element i: M = (lane>>4)*8 + i, N = lane&15
__device__ __forceinline__ void store_tile(h16* tile, int stride, int lane,
                                           v8f d, bool do_elu) {
  int n = lane & 15, mb = (lane >> 4) << 3;
#pragma unroll
  for (int i = 0; i < 8; ++i) {
    float v = d[i];
    if (do_elu) v = eluf(v);
    tile[(mb + i) * stride + n] = (h16)v;
  }
}

// ---------------------------------------------------------------------------
// Prep kernel: pack weights to f16 B-fragment layout; precompute pe_term.
// ---------------------------------------------------------------------------
__global__ __launch_bounds__(256) void ame2_prep_kernel(
    const float* __restrict__ conv1_w, const float* __restrict__ conv2_w,
    const float* __restrict__ fuse_w, const float* __restrict__ g_w1,
    const float* __restrict__ g_w2, const float* __restrict__ wk,
    const float* __restrict__ wv, const float* __restrict__ pe_w1,
    const float* __restrict__ pe_b1, const float* __restrict__ pe_w2,
    const float* __restrict__ pe_b2, const float* __restrict__ fuse_b,
    h16* __restrict__ wpack, float* __restrict__ pe_term) {
  const int t = threadIdx.x;
  // ---- weight packing ----
  for (int i = t; i < WPACK_HALVES; i += 256) {
    int li, nkt;
    if (i < OFF_C2)      { li = i - OFF_C1; nkt = 1; }
    else if (i < OFF_FU) { li = i - OFF_C2; nkt = 2; }
    else if (i < OFF_G1) { li = i - OFF_FU; nkt = 2; }
    else if (i < OFF_G2) { li = i - OFF_G1; nkt = 2; }
    else if (i < OFF_WK) { li = i - OFF_G2; nkt = 4; }
    else if (i < OFF_WV) { li = i - OFF_WK; nkt = 2; }
    else                 { li = i - OFF_WV; nkt = 2; }
    int e = li & 15, lane = (li >> 4) & 31, f = li >> 9;
    int kt = f % nkt, nt = f / nkt;
    int k = kt * 32 + ((lane >> 4) << 3) + (e & 7) + ((e >> 3) << 4);
    int n = nt * 16 + (lane & 15);
    float v = 0.f;
    if (i < OFF_C2) {
      if (k < 27) {  // im2col order: k = c*9 + ky*3 + kx ; OIHW source
        int c = k / 9, rem = k % 9, ky = rem / 3, kx = rem % 3;
        v = conv1_w[((n * 3 + c) * 3 + ky) * 3 + kx];
      }
    } else if (i < OFF_FU) v = conv2_w[n * 64 + k];      // (O,I,1,1) -> B[k][n]
    else if (i < OFF_G1)   v = fuse_w[k * 64 + n];       // (in,out), top half
    else if (i < OFF_G2)   v = g_w1[k * 128 + n];
    else if (i < OFF_WK)   v = g_w2[k * 128 + n];
    else if (i < OFF_WV)   v = wk[k * 64 + n];
    else                   v = wv[k * 64 + n];
    wpack[i] = (h16)v;
  }
  // ---- positional term: pe_term[l][n] = fuse_b[n] + sum_j pe[l][j]*fuse_w[64+j][n]
  __shared__ float h1[4][64];
  __shared__ float pr[4][64];
  const int r = t >> 6, n = t & 63;
  for (int l0 = 0; l0 < LP; l0 += 4) {
    int l = l0 + r;
    bool valid = l < LL;
    float gx = 0.f, gy = 0.f;
    if (valid) {
      int hh = l / WWD, ww = l - hh * WWD;
      gy = -1.f + (2.f / 13.f) * (float)hh;   // linspace(-1,1,14)
      gx = -1.f + (2.f / 35.f) * (float)ww;   // linspace(-1,1,36)
    }
    __syncthreads();
    h1[r][n] = eluf(gx * pe_w1[n] + gy * pe_w1[64 + n] + pe_b1[n]);
    __syncthreads();
    float p = pe_b2[n];
#pragma unroll 8
    for (int j = 0; j < 64; ++j) p += h1[r][j] * pe_w2[j * 64 + n];
    pr[r][n] = p;
    __syncthreads();
    float o = fuse_b[n];
#pragma unroll 8
    for (int j = 0; j < 64; ++j) o += pr[r][j] * fuse_w[(64 + j) * 64 + n];
    pe_term[l * 64 + n] = valid ? o : 0.f;
  }
}

// ---------------------------------------------------------------------------
// Main fused kernel: one workgroup (8 wave32) per batch element.
// ---------------------------------------------------------------------------
__global__ __launch_bounds__(256) void ame2_fused_kernel(
    const float* __restrict__ map_feat, const float* __restrict__ prop_emb,
    const float* __restrict__ conv1_b, const float* __restrict__ conv2_b,
    const float* __restrict__ g_b1, const float* __restrict__ g_b2,
    const float* __restrict__ bk, const float* __restrict__ bv,
    const float* __restrict__ qp_w, const float* __restrict__ qp_b,
    const float* __restrict__ wq, const float* __restrict__ bq,
    const float* __restrict__ wo, const float* __restrict__ bo,
    const float* __restrict__ pe_term, const h16* __restrict__ wpack,
    float* __restrict__ out) {
  __shared__ __align__(16) h16 sW[WPACK_HALVES];  // 84 KB weights (f16)
  __shared__ __align__(16) h16 sImc[32 * 32];     // im2col tile
  __shared__ __align__(16) h16 sAct1[32 * 64];
  __shared__ __align__(16) h16 sAct2[32 * 64];
  __shared__ __align__(16) h16 sPw[32 * 64];
  __shared__ __align__(16) h16 sHid[32 * 128];
  __shared__ __align__(16) h16 sK[LP * 64];       // 64 KB
  __shared__ __align__(16) h16 sV[LP * 64];       // 64 KB
  __shared__ float sGvec[128];
  __shared__ float sQ[64];
  __shared__ float sQv[64];
  __shared__ float sCtx[64];

  const int b = blockIdx.x;
  const int tid = threadIdx.x;
  const int lane = tid & 31;
  const int wid = tid >> 5;

  // warm this block's map_feat slice into cache (global_prefetch_b8)
  {
    const char* mbase = (const char*)(map_feat + (size_t)b * 3 * HH * WWD);
    if (tid < (3 * HH * WWD * 4 + 127) / 128)
      __builtin_prefetch(mbase + tid * 128, 0, 3);
  }

  // ---- stage packed weights into LDS ----
#ifdef USE_TDM
  // Tensor Data Mover: one contiguous 84KB global->LDS DMA issued by wave 0.
  if (wid == 0) {
    const unsigned long long ga = (unsigned long long)(uintptr_t)wpack;
    const unsigned int lds_off = (unsigned int)(uintptr_t)&sW[0];
    const int n8 = WPACK_BYTES / 8;  // 8-byte elements
    // D# group 0: count=1 | lds_addr | global_addr[56:0] | type=2
    uint32x4 g0;
    g0[0] = 1u;                                   // count=1 (valid), user mode
    g0[1] = lds_off;                              // lds_addr (bytes)
    g0[2] = (unsigned int)(ga & 0xFFFFFFFFu);     // global_addr[31:0]
    g0[3] = (unsigned int)((ga >> 32) & 0x1FFFFFFu) | (2u << 30);  // type=2
    // D# group 1: data_size=8B, 1-D tile: tensor_dim0=tile_dim0=n8
    int32x8 g1;
    g1[0] = 3 << 16;                 // workgroup_mask=0, data_size=3 (8B)
    g1[1] = (n8 & 0xFFFF) << 16;     // tensor_dim0[15:0]
    g1[2] = ((n8 >> 16) & 0xFFFF) | (1 << 16);  // tensor_dim0[31:16], tensor_dim1=1
    g1[3] = n8 << 16;                // tile_dim0 = n8 (fits 16 bits)
    g1[4] = 1;                       // tile_dim1 = 1, tile_dim2 = 0 (2-D)
    g1[5] = n8;                      // tensor_dim0_stride[31:0]
    g1[6] = 0;                       // stride hi, tensor_dim1_stride lo
    g1[7] = 0;
    int32x4 gz = {0, 0, 0, 0};
#if __clang_major__ >= 23
    int32x8 gz8 = {0, 0, 0, 0, 0, 0, 0, 0};
    __builtin_amdgcn_tensor_load_to_lds(g0, g1, gz, gz, gz8, 0);
#else
    __builtin_amdgcn_tensor_load_to_lds(g0, g1, gz, gz, 0);
#endif
    __builtin_amdgcn_s_wait_tensorcnt(0);
  }
#else
  {
    const uint4* src = (const uint4*)wpack;
    uint4* dst = (uint4*)sW;
    for (int i = tid; i < WPACK_HALVES / 8; i += 256) dst[i] = src[i];
  }
#endif
  v8f regmax;
#pragma unroll
  for (int i = 0; i < 8; ++i) regmax[i] = -1e30f;
  __syncthreads();

#pragma unroll 1
  for (int it = 0; it < 16; ++it) {
    const int r0 = it * 32;
    // ---- im2col (32 rows x K=32, K padded past 27) ----
    for (int idx = tid; idx < 32 * 32; idx += 256) {
      int row = idx >> 5, k = idx & 31;
      int gr = r0 + row;
      float v = 0.f;
      if (gr < LL && k < 27) {
        int hh = gr / WWD, ww = gr - hh * WWD;
        int c = k / 9, rem = k % 9, ky = rem / 3, kx = rem % 3;
        int ih = hh + ky - 1, iw = ww + kx - 1;
        if ((unsigned)ih < (unsigned)HH && (unsigned)iw < (unsigned)WWD)
          v = map_feat[((b * 3 + c) * HH + ih) * WWD + iw];
      }
      sImc[idx] = (h16)v;
    }
    __syncthreads();
    // ---- conv1: (32x32)x(32x64), 8 waves = 2 Mt x 4 Nt ----
    {
      int mt = wid >> 2, nt = wid & 3;
      v16h a = load_Afrag(sImc + mt * 16 * 32, 32, lane);
      v16h bb = load_Bfrag(sW + OFF_C1, nt, 1, 0, lane);
      v8f acc = bias_C(conv1_b, nt * 16, lane);
      acc = wmma32(a, bb, acc);
      store_tile(sAct1 + mt * 16 * 64 + nt * 16, 64, lane, acc, true);
    }
    __syncthreads();
    // ---- conv2 1x1: (32x64)x(64x64) ----
    {
      int mt = wid >> 2, nt = wid & 3;
      v8f acc = bias_C(conv2_b, nt * 16, lane);
#pragma unroll
      for (int kt = 0; kt < 2; ++kt) {
        v16h a = load_Afrag(sAct1 + mt * 16 * 64 + kt * 32, 64, lane);
        acc = wmma32(a, load_Bfrag(sW + OFF_C2, nt, 2, kt, lane), acc);
      }
      store_tile(sAct2 + mt * 16 * 64 + nt * 16, 64, lane, acc, true);
    }
    __syncthreads();
    // ---- fuse: (32x64)x(64x64) with C = pe_term tile (bias folded in) ----
    {
      int mt = wid >> 2, nt = wid & 3;
      int n = nt * 16 + (lane & 15), mb = (lane >> 4) << 3;
      v8f acc;
#pragma unroll
      for (int i = 0; i < 8; ++i)
        acc[i] = pe_term[(r0 + mt * 16 + mb + i) * 64 + n];
#pragma unroll
      for (int kt = 0; kt < 2; ++kt) {
        v16h a = load_Afrag(sAct2 + mt * 16 * 64 + kt * 32, 64, lane);
        acc = wmma32(a, load_Bfrag(sW + OFF_FU, nt, 2, kt, lane), acc);
      }
      store_tile(sPw + mt * 16 * 64 + nt * 16, 64, lane, acc, true);
    }
    __syncthreads();
    // ---- g1 (wave = Nt of 8, loop Mt) + K/V projections (2Mt x 4Nt) ----
    {
#pragma unroll
      for (int mt = 0; mt < 2; ++mt) {
        v8f acc = bias_C(g_b1, wid * 16, lane);
#pragma unroll
        for (int kt = 0; kt < 2; ++kt) {
          v16h a = load_Afrag(sPw + mt * 16 * 64 + kt * 32, 64, lane);
          acc = wmma32(a, load_Bfrag(sW + OFF_G1, wid, 2, kt, lane), acc);
        }
        store_tile(sHid + mt * 16 * 128 + wid * 16, 128, lane, acc, true);
      }
      int mt = wid >> 2, nt = wid & 3;
      v8f acck = bias_C(bk, nt * 16, lane);
      v8f accv = bias_C(bv, nt * 16, lane);
#pragma unroll
      for (int kt = 0; kt < 2; ++kt) {
        v16h a = load_Afrag(sPw + mt * 16 * 64 + kt * 32, 64, lane);
        acck = wmma32(a, load_Bfrag(sW + OFF_WK, nt, 2, kt, lane), acck);
        accv = wmma32(a, load_Bfrag(sW + OFF_WV, nt, 2, kt, lane), accv);
      }
      store_tile(sK + (r0 + mt * 16) * 64 + nt * 16, 64, lane, acck, false);
      store_tile(sV + (r0 + mt * 16) * 64 + nt * 16, 64, lane, accv, false);
    }
    __syncthreads();
    // ---- g2: (32x128)x(128x128) + masked running row-max ----
    {
#pragma unroll
      for (int mt = 0; mt < 2; ++mt) {
        v8f acc = bias_C(g_b2, wid * 16, lane);
#pragma unroll
        for (int kt = 0; kt < 4; ++kt) {
          v16h a = load_Afrag(sHid + mt * 16 * 128 + kt * 32, 128, lane);
          acc = wmma32(a, load_Bfrag(sW + OFF_G2, wid, 4, kt, lane), acc);
        }
        int mb = (lane >> 4) << 3;
#pragma unroll
        for (int i = 0; i < 8; ++i) {
          int row = r0 + mt * 16 + mb + i;
          if (row < LL) regmax[i] = fmaxf(regmax[i], acc[i]);
        }
      }
    }
    __syncthreads();
  }

  // ---- finish global max: wave 'wid' owns columns [wid*16, wid*16+16) ----
  {
    float m = -1e30f;
#pragma unroll
    for (int i = 0; i < 8; ++i) m = fmaxf(m, regmax[i]);
    m = fmaxf(m, __shfl_xor(m, 16, 32));  // merge M-halves (same N)
    if (lane < 16) sGvec[wid * 16 + lane] = m;
  }
  __syncthreads();

  // ---- q = elu([global_feat, prop_emb] @ qp_w + qp_b) ----
  if (tid < 64) {
    const float* pe = prop_emb + (size_t)b * 128;
    float a = qp_b[tid];
#pragma unroll 8
    for (int j = 0; j < 128; ++j) a += sGvec[j] * qp_w[j * 64 + tid];
#pragma unroll 8
    for (int j = 0; j < 128; ++j) a += pe[j] * qp_w[(128 + j) * 64 + tid];
    sQ[tid] = eluf(a);
  }
  __syncthreads();
  if (tid < 64) {
    float a = bq[tid];
#pragma unroll 8
    for (int j = 0; j < 64; ++j) a += sQ[j] * wq[j * 64 + tid];
    sQv[tid] = a;
  }
  __syncthreads();

  // ---- attention: 16 heads x HD=4, 16 lanes per head ----
  {
    int h = tid >> 4, sub = tid & 15;
    float q0 = sQv[h * 4 + 0], q1 = sQv[h * 4 + 1];
    float q2 = sQv[h * 4 + 2], q3 = sQv[h * 4 + 3];
    float s[32];
    float m = -1e30f;
#pragma unroll
    for (int i = 0; i < 32; ++i) {
      int l = sub + (i << 4);
      const h16* kp = sK + l * 64 + h * 4;
      float sc = 0.5f * (q0 * (float)kp[0] + q1 * (float)kp[1] +
                         q2 * (float)kp[2] + q3 * (float)kp[3]);
      if (l >= LL) sc = -1e30f;
      s[i] = sc;
      m = fmaxf(m, sc);
    }
#pragma unroll
    for (int o = 8; o >= 1; o >>= 1) m = fmaxf(m, __shfl_xor(m, o, 32));
    float sum = 0.f, c0 = 0.f, c1 = 0.f, c2 = 0.f, c3 = 0.f;
#pragma unroll
    for (int i = 0; i < 32; ++i) {
      int l = sub + (i << 4);
      float p = __expf(s[i] - m);
      sum += p;
      const h16* vp = sV + l * 64 + h * 4;
      c0 += p * (float)vp[0];
      c1 += p * (float)vp[1];
      c2 += p * (float)vp[2];
      c3 += p * (float)vp[3];
    }
#pragma unroll
    for (int o = 8; o >= 1; o >>= 1) {
      sum += __shfl_xor(sum, o, 32);
      c0 += __shfl_xor(c0, o, 32);
      c1 += __shfl_xor(c1, o, 32);
      c2 += __shfl_xor(c2, o, 32);
      c3 += __shfl_xor(c3, o, 32);
    }
    if (sub == 0) {
      float inv = 1.f / sum;
      sCtx[h * 4 + 0] = c0 * inv;
      sCtx[h * 4 + 1] = c1 * inv;
      sCtx[h * 4 + 2] = c2 * inv;
      sCtx[h * 4 + 3] = c3 * inv;
    }
  }
  __syncthreads();

  // ---- output: [ctx @ wo + bo , global_feat] ----
  if (tid < 64) {
    float a = bo[tid];
#pragma unroll 8
    for (int j = 0; j < 64; ++j) a += sCtx[j] * wo[j * 64 + tid];
    out[(size_t)b * 192 + tid] = a;
  }
  if (tid >= 64 && tid < 192) {
    out[(size_t)b * 192 + tid] = sGvec[tid - 64];
  }
}

extern "C" void kernel_launch(void* const* d_in, const int* in_sizes, int n_in,
                              void* d_out, int out_size, void* d_ws,
                              size_t ws_size, hipStream_t stream) {
  const float* map_feat = (const float*)d_in[0];
  const float* prop_emb = (const float*)d_in[1];
  const float* conv1_w = (const float*)d_in[2];
  const float* conv1_b = (const float*)d_in[3];
  const float* conv2_w = (const float*)d_in[4];
  const float* conv2_b = (const float*)d_in[5];
  const float* pe_w1 = (const float*)d_in[6];
  const float* pe_b1 = (const float*)d_in[7];
  const float* pe_w2 = (const float*)d_in[8];
  const float* pe_b2 = (const float*)d_in[9];
  const float* fuse_w = (const float*)d_in[10];
  const float* fuse_b = (const float*)d_in[11];
  const float* g_w1 = (const float*)d_in[12];
  const float* g_b1 = (const float*)d_in[13];
  const float* g_w2 = (const float*)d_in[14];
  const float* g_b2 = (const float*)d_in[15];
  const float* qp_w = (const float*)d_in[16];
  const float* qp_b = (const float*)d_in[17];
  const float* wq = (const float*)d_in[18];
  const float* bq = (const float*)d_in[19];
  const float* wk = (const float*)d_in[20];
  const float* bk = (const float*)d_in[21];
  const float* wv = (const float*)d_in[22];
  const float* bv = (const float*)d_in[23];
  const float* wo = (const float*)d_in[24];
  const float* bo = (const float*)d_in[25];

  float* pe_term = (float*)d_ws;
  h16* wpack = (h16*)((char*)d_ws + PETERM_BYTES);
  float* out = (float*)d_out;

  const int B = in_sizes[1] / 128;  // prop_emb is (B, 128)

  ame2_prep_kernel<<<1, 256, 0, stream>>>(conv1_w, conv2_w, fuse_w, g_w1, g_w2,
                                          wk, wv, pe_w1, pe_b1, pe_w2, pe_b2,
                                          fuse_b, wpack, pe_term);
  ame2_fused_kernel<<<B, 256, 0, stream>>>(
      map_feat, prop_emb, conv1_b, conv2_b, g_b1, g_b2, bk, bv, qp_w, qp_b, wq,
      bq, wo, bo, pe_term, wpack, out);
}
